// Seq2SeqDecoder_65146063945748
// MI455X (gfx1250) — matrix operand order repaired and checked by reading, not compiled, not run on hardware
//
#include <hip/hip_runtime.h>

typedef unsigned short u16;
typedef unsigned int   u32;
typedef __attribute__((ext_vector_type(16))) __bf16 v16bf;
typedef __attribute__((ext_vector_type(8)))  float  v8f;

// ---------------- problem constants ----------------
#define BB   64
#define TT   1000
#define INP  80
#define INPP 96      // padded input K
#define HH   512
#define GG   2048    // 4*HH
#define VV   1000
#define MLEN 100
#define NWG_ENC 64

// ---------------- workspace layout (bytes) ----------------
static constexpr size_t aln(size_t x) { return (x + 255) & ~size_t(255); }
static constexpr size_t SZ_XPAD  = (size_t)TT * BB * INPP * 2;       // x, time-major, padded, bf16
static constexpr size_t SZ_ENC   = (size_t)TT * BB * HH * 2;         // enc_out [t][b][h] bf16
static constexpr size_t SZ_PACKE = (size_t)128 * 19 * 32 * 16 * 2;   // encoder packed W
static constexpr size_t SZ_PACKD = (size_t)128 * 16 * 32 * 16 * 2;   // decoder packed W (x2)
static constexpr size_t SZ_PACKA = (size_t)32 * 32 * 32 * 16 * 2;    // attn packed W
static constexpr size_t SZ_PACKF = (size_t)63 * 16 * 32 * 16 * 2;    // fc packed W (N padded 1008)
static constexpr size_t SZ_BIAS  = (size_t)GG * 4;
static constexpr size_t SZ_HBUF  = (size_t)BB * HH * 2;
static constexpr size_t SZ_C     = (size_t)BB * HH * 4;
static constexpr size_t SZ_CCAT  = (size_t)BB * 1024 * 2;
static constexpr size_t SZ_ATTO  = (size_t)BB * HH * 2;
static constexpr size_t SZ_AW    = (size_t)BB * TT * 4;

static constexpr size_t OFF_XPAD  = 0;
static constexpr size_t OFF_ENC   = OFF_XPAD  + aln(SZ_XPAD);
static constexpr size_t OFF_PACKE = OFF_ENC   + aln(SZ_ENC);
static constexpr size_t OFF_PACKDH= OFF_PACKE + aln(SZ_PACKE);
static constexpr size_t OFF_PACKDS= OFF_PACKDH+ aln(SZ_PACKD);
static constexpr size_t OFF_PACKA = OFF_PACKDS+ aln(SZ_PACKD);
static constexpr size_t OFF_PACKF = OFF_PACKA + aln(SZ_PACKA);
static constexpr size_t OFF_BE    = OFF_PACKF + aln(SZ_PACKF);
static constexpr size_t OFF_BD    = OFF_BE    + aln(SZ_BIAS);
static constexpr size_t OFF_HB0   = OFF_BD    + aln(SZ_BIAS);
static constexpr size_t OFF_HB1   = OFF_HB0   + aln(SZ_HBUF);
static constexpr size_t OFF_CD    = OFF_HB1   + aln(SZ_HBUF);
static constexpr size_t OFF_CCAT  = OFF_CD    + aln(SZ_C);
static constexpr size_t OFF_ATTO  = OFF_CCAT  + aln(SZ_CCAT);
static constexpr size_t OFF_AW    = OFF_ATTO  + aln(SZ_ATTO);
static constexpr size_t OFF_BAR   = OFF_AW    + aln(SZ_AW);

// ---------------- device helpers ----------------
__device__ __forceinline__ u16 f2bf(float f) {
  union { float f; u32 u; } v; v.f = f;
  u32 r = v.u + 0x7FFFu + ((v.u >> 16) & 1u);
  return (u16)(r >> 16);
}
__device__ __forceinline__ float bf2f(u16 h) {
  union { u32 u; float f; } v; v.u = ((u32)h) << 16;
  return v.f;
}
struct Q2 { uint4 a, b; };
__device__ __forceinline__ v16bf frag2(const u16* p0, const u16* p1) {
  Q2 q; q.a = *(const uint4*)p0; q.b = *(const uint4*)p1;
  return __builtin_bit_cast(v16bf, q);
}
__device__ __forceinline__ v16bf frag32(const u16* p) {
  Q2 q; q.a = ((const uint4*)p)[0]; q.b = ((const uint4*)p)[1];
  return __builtin_bit_cast(v16bf, q);
}
__device__ __forceinline__ v8f wmma_bf16(v16bf a, v16bf b, v8f c) {
  return __builtin_amdgcn_wmma_f32_16x16x32_bf16(false, a, false, b, (short)0, c, false, false);
}
__device__ __forceinline__ float sigm(float x) { return 1.f / (1.f + __expf(-x)); }

// ---------------- prep kernels ----------------
__global__ void __launch_bounds__(256) k_init(u32* hb0, u32* hb1, u32* bar) {
  int i = blockIdx.x * 256 + threadIdx.x;
  if (i < (BB * HH) / 2) { hb0[i] = 0u; hb1[i] = 0u; }
  if (i == 0) bar[0] = 0u;
}

__global__ void __launch_bounds__(256) k_pad_x(const float* __restrict__ x, u16* __restrict__ xpad) {
  int idx = blockIdx.x * 256 + threadIdx.x;          // over T*B*INPP
  if (idx >= TT * BB * INPP) return;
  int k = idx % INPP, bt = idx / INPP;
  int b = bt % BB, t = bt / BB;
  float v = (k < INP) ? x[((size_t)b * TT + t) * INP + k] : 0.f;
  xpad[idx] = f2bf(v);
}

__global__ void __launch_bounds__(256) k_bias(const float* a0, const float* a1, float* be,
                                              const float* d0, const float* d1, float* bd) {
  int i = blockIdx.x * 256 + threadIdx.x;
  if (i < GG) { be[i] = a0[i] + a1[i]; bd[i] = d0[i] + d1[i]; }
}

// Pack [N][K] fp32 weight(s) into WMMA B-fragment tiles (bf16).
// K space = [0,KaSpan) from Wa (Ksrc=KaSrc, zero padded) then Wb (+optional Wb2) with KbSrc.
__global__ void __launch_bounds__(256) k_pack(u16* __restrict__ dst,
    const float* __restrict__ Wa, int KaSrc, int KaSpan,
    const float* __restrict__ Wb, const float* __restrict__ Wb2, int KbSrc,
    int Nsrc, int ntiles, int kblocks) {
  int idx = blockIdx.x * 256 + threadIdx.x;
  int total = ntiles * kblocks * 32;
  if (idx >= total) return;
  int lane = idx & 31;
  int kb = (idx >> 5) % kblocks;
  int nt = (idx >> 5) / kblocks;
  int n = nt * 16 + (lane & 15);
  int kbase = kb * 32 + (lane >> 4) * 8;
  u16 vals[16];
#pragma unroll
  for (int j = 0; j < 16; ++j) {
    int k = kbase + j + ((j >= 8) ? 8 : 0);
    float v = 0.f;
    if (n < Nsrc) {
      if (k < KaSpan) {
        if (k < KaSrc) v = Wa[(size_t)n * KaSrc + k];
      } else {
        int kk = k - KaSpan;
        if (kk < KbSrc) {
          v = Wb[(size_t)n * KbSrc + kk];
          if (Wb2) v += Wb2[(size_t)n * KbSrc + kk];
        }
      }
    }
    vals[j] = f2bf(v);
  }
  uint4* d = (uint4*)(dst + (size_t)idx * 16);
  d[0] = ((const uint4*)vals)[0];
  d[1] = ((const uint4*)vals)[1];
}

// ---------------- encoder: persistent, grid-synced, 1000 steps ----------------
__global__ void __launch_bounds__(256) k_encoder(
    const u16* __restrict__ xpad, const u16* __restrict__ packW,
    const float* __restrict__ bias, u16* hb0, u16* hb1,
    u16* __restrict__ enc, float* __restrict__ c_out, u32* bar) {
  __shared__ u16  Ax[32 * INPP];
  __shared__ u16  Ah[32 * HH];
  __shared__ float gat[32 * 64];
  __shared__ float cbuf[32 * 16];
  const int tid = threadIdx.x;
  const int wg = blockIdx.x;
  const int s = wg & 31;        // hidden-unit slice (16 units)
  const int half = wg >> 5;     // batch half (32 rows)
  const int lane = tid & 31, wave = tid >> 5;
  const int mt = wave >> 2;     // local M tile (0..1)
  const int gidx = wave & 3;    // gate index i,f,g,o
  const int ntile = gidx * 32 + s;             // global gate N-tile 0..127
  const float bv = bias[ntile * 16 + (lane & 15)];
  const int mloc = mt * 16 + (lane & 15);
  const int kb8 = (lane >> 4) * 8;
  u16* hb[2] = { hb0, hb1 };

  for (int i = tid; i < 32 * 16; i += 256) cbuf[i] = 0.f;

  for (int t = 0; t < TT; ++t) {
    const u16* hr = hb[t & 1];
    u16* hw = hb[(t & 1) ^ 1];
    // stage A operands into LDS
    {
      const uint4* sx = (const uint4*)(xpad + ((size_t)t * BB + half * 32) * INPP);
      uint4* dx = (uint4*)Ax;
      for (int i = tid; i < (32 * INPP) / 8; i += 256) dx[i] = sx[i];
      const uint4* sh = (const uint4*)(hr + (size_t)(half * 32) * HH);
      uint4* dh = (uint4*)Ah;
      for (int i = tid; i < (32 * HH) / 8; i += 256) dh[i] = sh[i];
    }
    __syncthreads();

    v8f acc;
#pragma unroll
    for (int i = 0; i < 8; ++i) acc[i] = bv;

    const u16* bp = packW + (size_t)(ntile * 19) * 32 * 16 + (size_t)lane * 16;
#pragma unroll
    for (int kb = 0; kb < 3; ++kb) {          // x part (K 0..95)
      int k = kb * 32 + kb8;
      v16bf a = frag2(Ax + mloc * INPP + k, Ax + mloc * INPP + k + 16);
      v16bf b = frag32(bp + (size_t)kb * 32 * 16);
      acc = wmma_bf16(a, b, acc);
    }
#pragma unroll 4
    for (int kb = 3; kb < 19; ++kb) {         // h part (K 96..607)
      int k = (kb - 3) * 32 + kb8;
      __builtin_prefetch(bp + (size_t)(kb + 2) * 32 * 16, 0, 1);
      v16bf a = frag2(Ah + mloc * HH + k, Ah + mloc * HH + k + 16);
      v16bf b = frag32(bp + (size_t)kb * 32 * 16);
      acc = wmma_bf16(a, b, acc);
    }
    // stage gates tile to LDS: local [32 rows][64 cols: i|f|g|o]
    {
      int n = gidx * 16 + (lane & 15);
      int mrow = mt * 16 + (lane >> 4) * 8;
#pragma unroll
      for (int r = 0; r < 8; ++r) gat[(mrow + r) * 64 + n] = acc[r];
    }
    __syncthreads();
    // elementwise LSTM cell for owned (32 batch x 16 units)
#pragma unroll
    for (int it = 0; it < 2; ++it) {
      int idx = tid + it * 256;
      int u = idx & 15, mb = idx >> 4;
      float ii = gat[mb * 64 + u];
      float ff = gat[mb * 64 + 16 + u];
      float gv = gat[mb * 64 + 32 + u];
      float oo = gat[mb * 64 + 48 + u];
      float c0 = cbuf[mb * 16 + u];
      float cn = sigm(ff) * c0 + sigm(ii) * tanhf(gv);
      float hn = sigm(oo) * tanhf(cn);
      cbuf[mb * 16 + u] = cn;
      int b = half * 32 + mb;
      int hid = s * 16 + u;
      u16 hv = f2bf(hn);
      hw[b * HH + hid] = hv;
      enc[((size_t)t * BB + b) * HH + hid] = hv;
      if (t == TT - 1) c_out[b * HH + hid] = cn;
    }
    // grid barrier (release h writes, acquire everyone's)
    __threadfence();
    __syncthreads();
    if (tid == 0) {
      atomicAdd(bar, 1u);
      u32 target = (u32)NWG_ENC * (u32)(t + 1);
      while (*((volatile u32*)bar) < target) __builtin_amdgcn_s_sleep(2);
    }
    __syncthreads();
    __threadfence();
  }
}

// ---------------- decoder LSTM step (K=512, packed W = Whh_d or Wih_d+Whh_d) ----------------
__global__ void __launch_bounds__(256) k_declstm(
    const u16* __restrict__ hr, u16* __restrict__ hw,
    const u16* __restrict__ packW, const float* __restrict__ bias,
    float* __restrict__ cst, u16* __restrict__ ccat) {
  __shared__ float gat[32 * 64];
  const int tid = threadIdx.x;
  const int wg = blockIdx.x;
  const int s = wg & 31, half = wg >> 5;
  const int lane = tid & 31, wave = tid >> 5;
  const int mt = wave >> 2, gidx = wave & 3;
  const int ntile = gidx * 32 + s;
  const float bv = bias[ntile * 16 + (lane & 15)];
  const int mrowg = half * 32 + mt * 16 + (lane & 15);
  const int kb8 = (lane >> 4) * 8;

  v8f acc;
#pragma unroll
  for (int i = 0; i < 8; ++i) acc[i] = bv;
  const u16* bp = packW + (size_t)(ntile * 16) * 32 * 16 + (size_t)lane * 16;
#pragma unroll 4
  for (int kb = 0; kb < 16; ++kb) {
    int k = kb * 32 + kb8;
    v16bf a = frag2(hr + (size_t)mrowg * HH + k, hr + (size_t)mrowg * HH + k + 16);
    v16bf b = frag32(bp + (size_t)kb * 32 * 16);
    acc = wmma_bf16(a, b, acc);
  }
  {
    int n = gidx * 16 + (lane & 15);
    int mrow = mt * 16 + (lane >> 4) * 8;
#pragma unroll
    for (int r = 0; r < 8; ++r) gat[(mrow + r) * 64 + n] = acc[r];
  }
  __syncthreads();
#pragma unroll
  for (int it = 0; it < 2; ++it) {
    int idx = tid + it * 256;
    int u = idx & 15, mb = idx >> 4;
    float ii = gat[mb * 64 + u];
    float ff = gat[mb * 64 + 16 + u];
    float gv = gat[mb * 64 + 32 + u];
    float oo = gat[mb * 64 + 48 + u];
    int b = half * 32 + mb;
    int hid = s * 16 + u;
    float c0 = cst[b * HH + hid];
    float cn = sigm(ff) * c0 + sigm(ii) * tanhf(gv);
    float hn = sigm(oo) * tanhf(cn);
    cst[b * HH + hid] = cn;
    u16 hv = f2bf(hn);
    hw[b * HH + hid] = hv;
    ccat[b * 1024 + HH + hid] = hv;   // second half of [ctx|out]
  }
}

// ---------------- attention scores + softmax (one WG per batch row) ----------------
__global__ void __launch_bounds__(256) k_scores(
    const u16* __restrict__ ccat, const u16* __restrict__ enc, float* __restrict__ aw) {
  __shared__ float outv[HH];
  __shared__ float sc[TT];
  __shared__ float red[8];
  const int b = blockIdx.x;
  const int tid = threadIdx.x, lane = tid & 31, wave = tid >> 5;
  for (int i = tid; i < HH; i += 256) outv[i] = bf2f(ccat[b * 1024 + HH + i]);
  __syncthreads();
  for (int t = wave; t < TT; t += 8) {
    const u16* ep = enc + ((size_t)t * BB + b) * HH;
    float sum = 0.f;
#pragma unroll
    for (int seg = 0; seg < 4; ++seg) {
      int h = seg * 128 + lane * 4;
      uint2 q = *(const uint2*)(ep + h);
      sum += bf2f((u16)(q.x & 0xffff)) * outv[h]
           + bf2f((u16)(q.x >> 16))    * outv[h + 1]
           + bf2f((u16)(q.y & 0xffff)) * outv[h + 2]
           + bf2f((u16)(q.y >> 16))    * outv[h + 3];
    }
#pragma unroll
    for (int off = 16; off > 0; off >>= 1) sum += __shfl_xor(sum, off, 32);
    if (lane == 0) sc[t] = sum;
  }
  __syncthreads();
  float m = -1e30f;
  for (int i = tid; i < TT; i += 256) m = fmaxf(m, sc[i]);
#pragma unroll
  for (int off = 16; off > 0; off >>= 1) m = fmaxf(m, __shfl_xor(m, off, 32));
  if (lane == 0) red[wave] = m;
  __syncthreads();
  m = red[0];
#pragma unroll
  for (int w = 1; w < 8; ++w) m = fmaxf(m, red[w]);
  __syncthreads();
  float ssum = 0.f;
  for (int i = tid; i < TT; i += 256) { float e = __expf(sc[i] - m); sc[i] = e; ssum += e; }
#pragma unroll
  for (int off = 16; off > 0; off >>= 1) ssum += __shfl_xor(ssum, off, 32);
  if (lane == 0) red[wave] = ssum;
  __syncthreads();
  ssum = 0.f;
#pragma unroll
  for (int w = 0; w < 8; ++w) ssum += red[w];
  float inv = 1.f / ssum;
  for (int i = tid; i < TT; i += 256) aw[(size_t)b * TT + i] = sc[i] * inv;
}

// ---------------- context vector (one WG per batch row) ----------------
__global__ void __launch_bounds__(256) k_ctx(
    const float* __restrict__ aw, const u16* __restrict__ enc, u16* __restrict__ ccat) {
  const int b = blockIdx.x, tid = threadIdx.x;
  float a0 = 0.f, a1 = 0.f;
  const u16* ep = enc + (size_t)b * HH + tid * 2;
  const float* ap = aw + (size_t)b * TT;
  for (int t = 0; t < TT; ++t) {
    float w = ap[t];
    u32 q = *(const u32*)(ep + (size_t)t * BB * HH);
    a0 += w * bf2f((u16)(q & 0xffff));
    a1 += w * bf2f((u16)(q >> 16));
  }
  ccat[b * 1024 + tid * 2] = f2bf(a0);
  ccat[b * 1024 + tid * 2 + 1] = f2bf(a1);
}

// ---------------- attn linear: [64x1024] @ [1024x512] ----------------
__global__ void __launch_bounds__(256) k_gemm_attn(
    const u16* __restrict__ ccat, const u16* __restrict__ pack,
    const float* __restrict__ bias, u16* __restrict__ outb) {
  const int tid = threadIdx.x, lane = tid & 31, wave = tid >> 5;
  const int tile = blockIdx.x * 8 + wave;   // 0..127
  const int nt = tile & 31, mtile = tile >> 5;
  const float bv = bias[nt * 16 + (lane & 15)];
  const int m = mtile * 16 + (lane & 15);
  const int kb8 = (lane >> 4) * 8;
  v8f acc;
#pragma unroll
  for (int i = 0; i < 8; ++i) acc[i] = bv;
  const u16* bp = pack + (size_t)(nt * 32) * 32 * 16 + (size_t)lane * 16;
#pragma unroll 4
  for (int kb = 0; kb < 32; ++kb) {
    int k = kb * 32 + kb8;
    v16bf a = frag2(ccat + (size_t)m * 1024 + k, ccat + (size_t)m * 1024 + k + 16);
    v16bf b = frag32(bp + (size_t)kb * 32 * 16);
    acc = wmma_bf16(a, b, acc);
  }
  int mrow = mtile * 16 + (lane >> 4) * 8;
  int n = nt * 16 + (lane & 15);
#pragma unroll
  for (int r = 0; r < 8; ++r) outb[(size_t)(mrow + r) * HH + n] = f2bf(acc[r]);
}

// ---------------- fc: [64x512] @ [512x1000] -> d_out[:, step, :] ----------------
__global__ void __launch_bounds__(256) k_gemm_fc(
    const u16* __restrict__ A, const u16* __restrict__ pack,
    const float* __restrict__ bias, float* __restrict__ out, int step) {
  const int tid = threadIdx.x, lane = tid & 31, wave = tid >> 5;
  const int tile = blockIdx.x * 8 + wave;   // 0..255, 252 used
  if (tile < 252) {
    const int nt = tile % 63, mtile = tile / 63;
    const int n = nt * 16 + (lane & 15);
    const float bv = (n < VV) ? bias[n] : 0.f;
    const int m = mtile * 16 + (lane & 15);
    const int kb8 = (lane >> 4) * 8;
    v8f acc;
#pragma unroll
    for (int i = 0; i < 8; ++i) acc[i] = bv;
    const u16* bp = pack + (size_t)(nt * 16) * 32 * 16 + (size_t)lane * 16;
#pragma unroll 4
    for (int kb = 0; kb < 16; ++kb) {
      int k = kb * 32 + kb8;
      v16bf a = frag2(A + (size_t)m * HH + k, A + (size_t)m * HH + k + 16);
      v16bf b = frag32(bp + (size_t)kb * 32 * 16);
      acc = wmma_bf16(a, b, acc);
    }
    if (n < VV) {
      int mrow = mtile * 16 + (lane >> 4) * 8;
#pragma unroll
      for (int r = 0; r < 8; ++r)
        out[((size_t)(mrow + r) * MLEN + step) * VV + n] = acc[r];
    }
  }
}

// ---------------- host side ----------------
extern "C" void kernel_launch(void* const* d_in, const int* in_sizes, int n_in,
                              void* d_out, int out_size, void* d_ws, size_t ws_size,
                              hipStream_t stream) {
  (void)in_sizes; (void)n_in; (void)out_size; (void)ws_size;
  const float* x     = (const float*)d_in[0];
  const float* Wihe  = (const float*)d_in[1];
  const float* Whhe  = (const float*)d_in[2];
  const float* bihe  = (const float*)d_in[3];
  const float* bhhe  = (const float*)d_in[4];
  const float* Wihd  = (const float*)d_in[5];
  const float* Whhd  = (const float*)d_in[6];
  const float* bihd  = (const float*)d_in[7];
  const float* bhhd  = (const float*)d_in[8];
  const float* Wattn = (const float*)d_in[9];
  const float* battn = (const float*)d_in[10];
  const float* Wfc   = (const float*)d_in[11];
  const float* bfc   = (const float*)d_in[12];

  char* ws = (char*)d_ws;
  u16*   xpad   = (u16*)(ws + OFF_XPAD);
  u16*   encb   = (u16*)(ws + OFF_ENC);
  u16*   packE  = (u16*)(ws + OFF_PACKE);
  u16*   packDH = (u16*)(ws + OFF_PACKDH);
  u16*   packDS = (u16*)(ws + OFF_PACKDS);
  u16*   packA  = (u16*)(ws + OFF_PACKA);
  u16*   packF  = (u16*)(ws + OFF_PACKF);
  float* be     = (float*)(ws + OFF_BE);
  float* bd     = (float*)(ws + OFF_BD);
  u16*   hb0    = (u16*)(ws + OFF_HB0);
  u16*   hb1    = (u16*)(ws + OFF_HB1);
  float* c_d    = (float*)(ws + OFF_CD);
  u16*   ccat   = (u16*)(ws + OFF_CCAT);
  u16*   atto   = (u16*)(ws + OFF_ATTO);
  float* aw     = (float*)(ws + OFF_AW);
  u32*   bar    = (u32*)(ws + OFF_BAR);

  // prep
  k_init<<<((BB * HH / 2) + 255) / 256, 256, 0, stream>>>((u32*)hb0, (u32*)hb1, bar);
  k_pad_x<<<(TT * BB * INPP + 255) / 256, 256, 0, stream>>>(x, xpad);
  k_bias<<<(GG + 255) / 256, 256, 0, stream>>>(bihe, bhhe, be, bihd, bhhd, bd);
  k_pack<<<(128 * 19 * 32) / 256, 256, 0, stream>>>(packE, Wihe, INP, INPP, Whhe, nullptr, HH, GG, 128, 19);
  k_pack<<<(128 * 16 * 32) / 256, 256, 0, stream>>>(packDH, nullptr, 0, 0, Whhd, nullptr, HH, GG, 128, 16);
  k_pack<<<(128 * 16 * 32) / 256, 256, 0, stream>>>(packDS, nullptr, 0, 0, Whhd, Wihd, HH, GG, 128, 16);
  k_pack<<<(32 * 32 * 32) / 256, 256, 0, stream>>>(packA, nullptr, 0, 0, Wattn, nullptr, 1024, HH, 32, 32);
  k_pack<<<(63 * 16 * 32 + 255) / 256, 256, 0, stream>>>(packF, nullptr, 0, 0, Wfc, nullptr, HH, VV, 63, 16);

  // encoder: persistent, 1000 grid-synced steps
  k_encoder<<<NWG_ENC, 256, 0, stream>>>(xpad, packE, be, hb0, hb1, encb, c_d, bar);

  // decoder: 100 sequential steps (graph-captured launch chain)
  for (int s = 0; s < MLEN; ++s) {
    const int u = TT + s;                      // continue h double-buffer parity
    const u16* hr = (u & 1) ? hb1 : hb0;
    u16*       hw = (u & 1) ? hb0 : hb1;
    k_declstm<<<64, 256, 0, stream>>>(hr, hw, (s == 0) ? packDH : packDS, bd, c_d, ccat);
    k_scores<<<BB, 256, 0, stream>>>(ccat, encb, aw);
    k_ctx<<<BB, 256, 0, stream>>>(aw, encb, ccat);
    k_gemm_attn<<<16, 256, 0, stream>>>(ccat, packA, battn, atto);
    k_gemm_fc<<<32, 256, 0, stream>>>(atto, packF, bfc, (float*)d_out, s);
  }
}
